// VariationalDistribution_6021544149698
// MI455X (gfx1250) — matrix operand order repaired and checked by reading, not compiled, not run on hardware
//
#include <hip/hip_runtime.h>
#include <hip/hip_bf16.h>
#include <math.h>

// Problem dims
#define BSZ  128
#define TLEN 512
#define XD   64
#define HD   512
#define ZD   64

// Persistent grid: 36 WGs * 8 waves = 288 waves.
// 256 waves -> GRU / combiner-h tile tasks (8 M-tiles x 32 hidden-col tiles)
//  32 waves -> loc/scale tasks (8 M-tiles x 4 z-col tiles)
#define NWG 36
#define TPB 256

typedef __attribute__((ext_vector_type(16))) __bf16 v16bf;
typedef __attribute__((ext_vector_type(8)))  __bf16 v8bf;
typedef __attribute__((ext_vector_type(8)))  float  v8f;

// ---------------- WMMA fragment loaders (16x16x32 bf16) ----------------
// A (16x32, row major): lane L holds row m0+(L&15);
//   elems 0..7  = K[k0 + (L>>4)*8 .. +7],  elems 8..15 = K[k0+16+(L>>4)*8 .. +7]
__device__ __forceinline__ v16bf load_a(const __bf16* base, int lda, int m0, int k0, int lane) {
  const __bf16* p = base + (long)(m0 + (lane & 15)) * lda + k0 + ((lane >> 4) << 3);
  v8bf lo = *(const v8bf*)p;
  v8bf hi = *(const v8bf*)(p + 16);
  return __builtin_shufflevector(lo, hi, 0,1,2,3,4,5,6,7,8,9,10,11,12,13,14,15);
}
// B (32x16): column n = W row n (x@W^T). Lane L holds col (L&15),
//   elems 0..7  = K[k0 + (L>>4)*8 .. +7],  elems 8..15 = K[k0+16+(L>>4)*8 .. +7]
__device__ __forceinline__ v16bf load_b(const __bf16* wrow0, int ldb, int k0, int lane) {
  const __bf16* p = wrow0 + (long)(lane & 15) * ldb + k0 + ((lane >> 4) << 3);
  v8bf lo = *(const v8bf*)p;
  v8bf hi = *(const v8bf*)(p + 16);
  return __builtin_shufflevector(lo, hi, 0,1,2,3,4,5,6,7,8,9,10,11,12,13,14,15);
}

__device__ __forceinline__ v8f wmma_bf16(v16bf a, v16bf b, v8f c) {
  return __builtin_amdgcn_wmma_f32_16x16x32_bf16(false, a, false, b, (short)0, c,
                                                 false, false);
}

__device__ __forceinline__ v8f bcast8(float v) {
  v8f a;
#pragma unroll
  for (int i = 0; i < 8; ++i) a[i] = v;
  return a;
}

// Fast transcendentals on the HW TRANS pipe (v_exp/v_log/v_rcp).
__device__ __forceinline__ float fast_rcp(float x)  { return __builtin_amdgcn_rcpf(x); }
__device__ __forceinline__ float sigm(float x)      { return fast_rcp(1.0f + __expf(-x)); }
__device__ __forceinline__ float fast_tanh(float x) { return 1.0f - 2.0f * fast_rcp(1.0f + __expf(2.0f * x)); }
__device__ __forceinline__ float softplus(float x)  { return (x > 20.0f) ? x : __logf(1.0f + __expf(x)); }

// Monotonic-counter grid barrier (all WGs call it the same number of times).
__device__ __forceinline__ void grid_sync(unsigned* cnt, unsigned target) {
  __threadfence();
  __syncthreads();
  if (threadIdx.x == 0) {
    atomicAdd(cnt, 1u);
    while (atomicAdd(cnt, 0u) < target) { __builtin_amdgcn_s_sleep(1); }
    __threadfence();
  }
  __syncthreads();
}

// ---------------- prep kernels ----------------
__global__ void cvt_bf16(const float* __restrict__ src, __bf16* __restrict__ dst, int n) {
  int i = blockIdx.x * blockDim.x + threadIdx.x;
  if (i < n) dst[i] = (__bf16)src[i];
}

__global__ void init_state(const float* __restrict__ h0, const float* __restrict__ z0,
                           float* __restrict__ H, __bf16* __restrict__ Hbf0,
                           __bf16* __restrict__ Zbf, unsigned* __restrict__ cnt) {
  int i = blockIdx.x * blockDim.x + threadIdx.x;
  if (i == 0) *cnt = 0u;
  if (i < BSZ * HD) { float v = h0[i & (HD - 1)]; H[i] = v; Hbf0[i] = (__bf16)v; }
  if (i < BSZ * ZD) { Zbf[i] = (__bf16)z0[i & (ZD - 1)]; }
}

// loc/scale task for step tt (A = hc), fused so each hc A-fragment feeds 2 WMMAs.
__device__ __forceinline__ void p3_task(int tt, int m0p, int n0, int lane, int col, int kh,
                                        const __bf16* __restrict__ hc,
                                        const __bf16* __restrict__ Wloc,
                                        const __bf16* __restrict__ Wsc,
                                        const float* __restrict__ b_loc,
                                        const float* __restrict__ b_sc,
                                        const float* __restrict__ eps,
                                        float* __restrict__ out,
                                        __bf16* __restrict__ Zbf, bool writeZ) {
  const __bf16* Wl = Wloc + (long)n0 * HD;
  const __bf16* Ws = Wsc + (long)n0 * HD;
  v8f lacc = bcast8(b_loc[n0 + col]);
  v8f sacc = bcast8(b_sc[n0 + col]);
#pragma unroll 2
  for (int k0 = 0; k0 < HD; k0 += 32) {
    v16bf a = load_a(hc, HD, m0p, k0, lane);
    lacc = wmma_bf16(a, load_b(Wl, HD, k0, lane), lacc);
    sacc = wmma_bf16(a, load_b(Ws, HD, k0, lane), sacc);
  }
#pragma unroll
  for (int r = 0; r < 8; ++r) {
    int row = m0p + r + 8 * kh;                         // batch index
    float e  = eps[((long)row * TLEN + tt) * ZD + n0 + col];
    float zv = lacc[r] + softplus(sacc[r]) * e;
    out[((long)row * TLEN + tt) * ZD + n0 + col] = zv;
    if (writeZ) Zbf[(long)row * ZD + n0 + col] = (__bf16)zv;
  }
}

// ---------------- persistent DMM kernel ----------------
__global__ __launch_bounds__(TPB, 1)
void dmm_persistent(const __bf16* __restrict__ Xbf,   // [B,T,XD] bf16
                    const __bf16* __restrict__ Wih,   // [3H,XD]
                    const __bf16* __restrict__ Whh,   // [3H,HD]
                    const __bf16* __restrict__ Wzh,   // [HD,ZD]
                    const __bf16* __restrict__ Wloc,  // [ZD,HD]
                    const __bf16* __restrict__ Wsc,   // [ZD,HD]
                    const float* __restrict__ b_ih, const float* __restrict__ b_hh,
                    const float* __restrict__ b_zh, const float* __restrict__ b_loc,
                    const float* __restrict__ b_sc,
                    const float* __restrict__ eps,    // [B,T,ZD]
                    float* __restrict__ H,            // [B,HD] fp32 state
                    __bf16* __restrict__ Hbf0, __bf16* __restrict__ Hbf1,
                    __bf16* __restrict__ Zbf,         // [B,ZD]
                    __bf16* __restrict__ hc,          // [B,HD]
                    unsigned* __restrict__ cnt,
                    float* __restrict__ out)          // [B,T,ZD]
{
  const int lane = threadIdx.x & 31;
  const int wave = threadIdx.x >> 5;
  const int w    = blockIdx.x * (TPB / 32) + wave;   // 0..287
  const int col  = lane & 15;
  const int kh   = lane >> 4;

  // task decode
  const int m0g = ((w & 255) >> 5) * 16;   // batch tile (GRU/P2 tasks, w<256)
  const int j0  = (w & 31) * 16;           // hidden-col tile
  const int p3i = w - 256;                 // loc/scale task id
  const int m0p = (p3i >> 2) * 16;
  const int n0  = (p3i & 3) * 16;

  // per-wave fixed B bases (gate rows of the PyTorch-layout weights)
  const __bf16* Wx_r = Wih + (long)j0 * XD;
  const __bf16* Wx_z = Wih + (long)(HD + j0) * XD;
  const __bf16* Wx_n = Wih + (long)(2 * HD + j0) * XD;
  const __bf16* Wh_r = Whh + (long)j0 * HD;
  const __bf16* Wh_z = Whh + (long)(HD + j0) * HD;
  const __bf16* Wh_n = Whh + (long)(2 * HD + j0) * HD;

  unsigned round = 0;

  for (int t = 0; t < TLEN; ++t) {
    const __bf16* Hr = (t & 1) ? Hbf1 : Hbf0;
    __bf16*       Hw = (t & 1) ? Hbf0 : Hbf1;

    // ---- Phase A: GRU step t (waves 0..255) || loc/scale for step t-1 (waves 256..287)
    if (w < 256) {
      const __bf16* Xt = Xbf + (long)t * XD;          // row stride TLEN*XD
      v8f racc  = bcast8(b_ih[j0 + col] + b_hh[j0 + col]);
      v8f zacc  = bcast8(b_ih[HD + j0 + col] + b_hh[HD + j0 + col]);
      v8f inacc = bcast8(b_ih[2 * HD + j0 + col]);
      v8f hnacc = bcast8(b_hh[2 * HD + j0 + col]);
      // x-projection: one A fragment feeds all three gates
#pragma unroll
      for (int k0 = 0; k0 < XD; k0 += 32) {
        v16bf a = load_a(Xt, TLEN * XD, m0g, k0, lane);
        racc  = wmma_bf16(a, load_b(Wx_r, XD, k0, lane), racc);
        zacc  = wmma_bf16(a, load_b(Wx_z, XD, k0, lane), zacc);
        inacc = wmma_bf16(a, load_b(Wx_n, XD, k0, lane), inacc);
      }
      // h-projection: one A fragment feeds all three gates
#pragma unroll 2
      for (int k0 = 0; k0 < HD; k0 += 32) {
        v16bf a = load_a(Hr, HD, m0g, k0, lane);
        racc  = wmma_bf16(a, load_b(Wh_r, HD, k0, lane), racc);
        zacc  = wmma_bf16(a, load_b(Wh_z, HD, k0, lane), zacc);
        hnacc = wmma_bf16(a, load_b(Wh_n, HD, k0, lane), hnacc);
      }
#pragma unroll
      for (int r = 0; r < 8; ++r) {
        int row = m0g + r + 8 * kh;
        float rr = sigm(racc[r]);
        float zz = sigm(zacc[r]);
        float nn = fast_tanh(inacc[r] + rr * hnacc[r]);
        float ho = H[(long)row * HD + j0 + col];
        float hn = (1.0f - zz) * nn + zz * ho;
        H[(long)row * HD + j0 + col]  = hn;
        Hw[(long)row * HD + j0 + col] = (__bf16)hn;
      }
    } else if (t > 0) {
      p3_task(t - 1, m0p, n0, lane, col, kh, hc, Wloc, Wsc, b_loc, b_sc, eps, out,
              Zbf, /*writeZ=*/true);
    }
    grid_sync(cnt, (unsigned)NWG * (++round));

    // ---- Phase B: combiner h_c for step t (waves 0..255)
    if (w < 256) {
      const __bf16* Wz = Wzh + (long)j0 * ZD;
      v8f acc = bcast8(b_zh[j0 + col]);
#pragma unroll
      for (int k0 = 0; k0 < ZD; k0 += 32) {
        v16bf a = load_a(Zbf, ZD, m0g, k0, lane);
        acc = wmma_bf16(a, load_b(Wz, ZD, k0, lane), acc);
      }
#pragma unroll
      for (int r = 0; r < 8; ++r) {
        int row = m0g + r + 8 * kh;
        float h  = H[(long)row * HD + j0 + col];
        float hv = 0.5f * (fast_tanh(acc[r]) + h);
        hc[(long)row * HD + j0 + col] = (__bf16)hv;
      }
    }
    grid_sync(cnt, (unsigned)NWG * (++round));
  }

  // ---- trailing loc/scale for t = TLEN-1 (Zbf no longer needed)
  if (w >= 256) {
    p3_task(TLEN - 1, m0p, n0, lane, col, kh, hc, Wloc, Wsc, b_loc, b_sc, eps, out,
            Zbf, /*writeZ=*/false);
  }
}

// ---------------- host launcher ----------------
extern "C" void kernel_launch(void* const* d_in, const int* in_sizes, int n_in,
                              void* d_out, int out_size, void* d_ws, size_t ws_size,
                              hipStream_t stream) {
  const float* mini_batch = (const float*)d_in[0];
  const float* eps     = (const float*)d_in[1];
  const float* W_ih    = (const float*)d_in[2];
  const float* W_hh    = (const float*)d_in[3];
  const float* b_ih    = (const float*)d_in[4];
  const float* b_hh    = (const float*)d_in[5];
  const float* W_zh    = (const float*)d_in[6];
  const float* b_zh    = (const float*)d_in[7];
  const float* W_loc   = (const float*)d_in[8];
  const float* b_loc   = (const float*)d_in[9];
  const float* W_scale = (const float*)d_in[10];
  const float* b_scale = (const float*)d_in[11];
  const float* z0      = (const float*)d_in[12];
  const float* h0      = (const float*)d_in[13];
  float* out = (float*)d_out;

  char* ws = (char*)d_ws;
  size_t off = 0;
  auto take = [&](size_t bytes) { char* p = ws + off; off = (off + bytes + 255) & ~(size_t)255; return p; };
  __bf16* Xbf  = (__bf16*)take((size_t)BSZ * TLEN * XD * 2);
  __bf16* Wihb = (__bf16*)take((size_t)3 * HD * XD * 2);
  __bf16* Whhb = (__bf16*)take((size_t)3 * HD * HD * 2);
  __bf16* Wzhb = (__bf16*)take((size_t)HD * ZD * 2);
  __bf16* Wlocb= (__bf16*)take((size_t)ZD * HD * 2);
  __bf16* Wscb = (__bf16*)take((size_t)ZD * HD * 2);
  float*  H    = (float*)take((size_t)BSZ * HD * 4);
  __bf16* Hbf0 = (__bf16*)take((size_t)BSZ * HD * 2);
  __bf16* Hbf1 = (__bf16*)take((size_t)BSZ * HD * 2);
  __bf16* Zbf  = (__bf16*)take((size_t)BSZ * ZD * 2);
  __bf16* hc   = (__bf16*)take((size_t)BSZ * HD * 2);
  unsigned* cnt = (unsigned*)take(256);

  auto cvt = [&](const float* s, __bf16* d, int n) {
    cvt_bf16<<<(n + 255) / 256, 256, 0, stream>>>(s, d, n);
  };
  cvt(mini_batch, Xbf, BSZ * TLEN * XD);
  cvt(W_ih, Wihb, 3 * HD * XD);
  cvt(W_hh, Whhb, 3 * HD * HD);
  cvt(W_zh, Wzhb, HD * ZD);
  cvt(W_loc, Wlocb, ZD * HD);
  cvt(W_scale, Wscb, ZD * HD);

  init_state<<<(BSZ * HD + 255) / 256, 256, 0, stream>>>(h0, z0, H, Hbf0, Zbf, cnt);

  dmm_persistent<<<NWG, TPB, 0, stream>>>(Xbf, Wihb, Whhb, Wzhb, Wlocb, Wscb,
                                          b_ih, b_hh, b_zh, b_loc, b_scale,
                                          eps, H, Hbf0, Hbf1, Zbf, hc, cnt, out);
}
